// conv_knrm_74655121539141
// MI455X (gfx1250) — matrix operand order repaired and checked by reading, not compiled, not run on hardware
//
#include <hip/hip_runtime.h>
#include <hip/hip_bf16.h>

// ---------------------------------------------------------------------------
// conv-KNRM for MI455X (gfx1250, wave32, WMMA + async global->LDS)
//   B=64, Lq=32, Ld=1024, D=300 (padded to 320), C=128, 11 RBF kernels
// ---------------------------------------------------------------------------

typedef __attribute__((ext_vector_type(16))) _Float16 v16h;
typedef __attribute__((ext_vector_type(8)))  float    v8f;

union Frag16 {
  uint4 u4[2];   // 32 bytes
  v16h  h;       // 16 halfs
};

__device__ __forceinline__ v8f wmma_f16(v16h a, v16h b, v8f c) {
  // (neg_a, A, neg_b, B, c_mod, C, reuse_a, reuse_b)
  return __builtin_amdgcn_wmma_f32_16x16x32_f16(false, a, false, b, (short)0, c,
                                                false, false);
}

__device__ __forceinline__ void async_wait_all() {
#if __has_builtin(__builtin_amdgcn_s_wait_asynccnt)
  __builtin_amdgcn_s_wait_asynccnt(0);
#else
  asm volatile("s_wait_asynccnt 0" ::: "memory");
#endif
}

#define B_  64
#define LQ  32
#define LD  1024
#define DP  320      // padded D
#define CC  128      // channels / feature dim
#define NK  11       // RBF kernels

// ---------------------------------------------------------------------------
// prep kernels
// ---------------------------------------------------------------------------
__global__ void zero_f32_kernel(float* __restrict__ p, int n) {
  int i = blockIdx.x * blockDim.x + threadIdx.x;
  if (i < n) p[i] = 0.0f;
}

// (B,L,300) f32 -> (B,Lalloc,320) f16, zero-padded rows/cols
__global__ void cvt_x_kernel(const float* __restrict__ src, _Float16* __restrict__ dst,
                             int L, int Lalloc) {
  size_t idx = (size_t)blockIdx.x * blockDim.x + threadIdx.x;
  size_t total = (size_t)B_ * Lalloc * DP;
  if (idx >= total) return;
  int col = (int)(idx % DP);
  int row = (int)((idx / DP) % Lalloc);
  int b   = (int)(idx / ((size_t)DP * Lalloc));
  float v = 0.0f;
  if (row < L && col < 300) v = src[((size_t)b * L + row) * 300 + col];
  dst[idx] = (_Float16)v;
}

// (C,k,300) f32 -> (C,k*320) f16, zero pads
__global__ void cvt_w_kernel(const float* __restrict__ src, _Float16* __restrict__ dst,
                             int k) {
  int idx = blockIdx.x * blockDim.x + threadIdx.x;
  int total = CC * k * DP;
  if (idx >= total) return;
  int col = idx % DP;
  int jc  = idx / DP;            // c*k + j
  float v = (col < 300) ? src[jc * 300 + col] : 0.0f;
  dst[idx] = (_Float16)v;
}

// ---------------------------------------------------------------------------
// n-gram conv + bias + relu + 1e-9 + l2norm  ->  f16 features (B, Lpad, 128)
// One workgroup = (batch, 16-row tile). 128 threads = 4 waves, each wave
// owns 32 channels (two 16x16 WMMA accumulators). A tile staged in LDS via
// CDNA5 async global->LDS DMA (ASYNCcnt).
// ---------------------------------------------------------------------------
__global__ __launch_bounds__(128) void ngram_conv_kernel(
    const _Float16* __restrict__ x16,   // (B, Lalloc, 320)
    const _Float16* __restrict__ w16,   // (128, K)   K = k*320
    const float*    __restrict__ bias,  // (128)
    _Float16*       __restrict__ feat,  // (B, Lpad, 128)
    int Lalloc, int Lpad, int K) {
  extern __shared__ _Float16 smem[];
  _Float16* smemA  = smem;                        // 16 x K halfs
  float*    rowsum = (float*)(smem + 16 * K);     // 16 floats

  const int b       = blockIdx.x;
  const int rowbase = blockIdx.y * 16;
  const int tid     = threadIdx.x;
  const int lane    = tid & 31;
  const int wid     = tid >> 5;
  const int l16     = lane & 15;
  const int hi      = lane >> 4;           // 0: lanes 0-15, 1: lanes 16-31

  // A tile staging: 16 overlapping k-gram rows (each K contiguous halfs),
  // streamed straight into LDS with global_load_async_to_lds_b128.
  const int per_row = K / 8;               // b128 items per row
  const int items   = 16 * per_row;
  for (int i = tid; i < items; i += 128) {
    int row = i / per_row;
    int off = (i % per_row) * 8;
    const _Float16* src = x16 + ((size_t)b * Lalloc + rowbase + row) * DP + off;
    unsigned ldsa = (unsigned)(uintptr_t)(smemA + row * K + off);
    asm volatile("global_load_async_to_lds_b128 %0, %1, off"
                 :: "v"(ldsa), "v"(src) : "memory");
  }
  if (tid < 16) rowsum[tid] = 0.0f;
  async_wait_all();
  __syncthreads();

  const int col0 = wid * 32 + l16;
  const int col1 = col0 + 16;

  v8f c0 = {}; v8f c1 = {};
  const int nchunks = K / 32;
  const _Float16* b0base = w16 + (size_t)col0 * K + hi * 16;
  const _Float16* b1base = w16 + (size_t)col1 * K + hi * 16;

  for (int ch = 0; ch < nchunks; ++ch) {
    Frag16 a, bf0, bf1;
    const _Float16* arp = smemA + l16 * K + ch * 32 + hi * 16;
    a.u4[0] = *(const uint4*)(arp);
    a.u4[1] = *(const uint4*)(arp + 8);
    const _Float16* b0p = b0base + ch * 32;
    const _Float16* b1p = b1base + ch * 32;
    bf0.u4[0] = ((const uint4*)b0p)[0];
    bf0.u4[1] = ((const uint4*)b0p)[1];
    bf1.u4[0] = ((const uint4*)b1p)[0];
    bf1.u4[1] = ((const uint4*)b1p)[1];
    if (ch + 1 < nchunks) {                       // global_prefetch_b8
      __builtin_prefetch(b0p + 32, 0, 0);
      __builtin_prefetch(b1p + 32, 0, 0);
    }
    c0 = wmma_f16(a.h, bf0.h, c0);
    c1 = wmma_f16(a.h, bf1.h, c1);
  }

  // bias + relu + eps; accumulate per-row squared sums across the WGP
  const float bias0 = bias[col0];
  const float bias1 = bias[col1];
#pragma unroll
  for (int v = 0; v < 8; ++v) {
    float t0 = fmaxf(c0[v] + bias0, 0.0f) + 1e-9f;
    float t1 = fmaxf(c1[v] + bias1, 0.0f) + 1e-9f;
    c0[v] = t0; c1[v] = t1;
    atomicAdd(&rowsum[v + hi * 8], t0 * t0 + t1 * t1);
  }
  __syncthreads();

#pragma unroll
  for (int v = 0; v < 8; ++v) {
    int M = v + hi * 8;
    float inv = 1.0f / fmaxf(sqrtf(rowsum[M]), 1e-10f);
    size_t base = ((size_t)b * Lpad + rowbase + M) * CC;
    feat[base + col0] = (_Float16)(c0[v] * inv);
    feat[base + col1] = (_Float16)(c1[v] * inv);
  }
}

// ---------------------------------------------------------------------------
// sim = q . d^T via WMMA, fused 11-kernel RBF pooling, log, masks
// grid: (64 batches, 9 combos, 2 q-tiles); block 128 = 4 waves,
// each wave sweeps 1024/64 doc-tiles strided by 4.
// ---------------------------------------------------------------------------
struct SimParams {
  const _Float16* qf[3];    // uni/bi/tri query features  (B,32,128)  f16
  const _Float16* df[3];    // uni/bi/tri doc   features  (B,1024,128) f16
  const float* qwm;         // (B,32)
  const float* dwm;         // (B,1024)
  float* pool;              // (B,99)
};

__global__ __launch_bounds__(128) void sim_pool_kernel(SimParams p) {
  static const float MU[NK] = {1.0f, 0.9f, 0.7f, 0.5f, 0.3f, 0.1f,
                               -0.1f, -0.3f, -0.5f, -0.7f, -0.9f};
  static const float NC[NK] = {-500000.0f, -50.0f, -50.0f, -50.0f, -50.0f, -50.0f,
                               -50.0f, -50.0f, -50.0f, -50.0f, -50.0f};
  static const int QSEL[9] = {0, 0, 0, 1, 2, 1, 1, 2, 2};
  static const int DSEL[9] = {0, 2, 1, 0, 0, 1, 2, 1, 2};

  const int b     = blockIdx.x;
  const int combo = blockIdx.y;
  const int qtile = blockIdx.z;
  const int qi = QSEL[combo], di = DSEL[combo];
  const _Float16* qf = p.qf[qi] + (size_t)b * LQ * CC;
  const _Float16* df = p.df[di] + (size_t)b * LD * CC;

  __shared__ float ps[16 * NK];
  for (int i = threadIdx.x; i < 16 * NK; i += 128) ps[i] = 0.0f;
  __syncthreads();

  const int lane = threadIdx.x & 31;
  const int wid  = threadIdx.x >> 5;
  const int l16  = lane & 15;
  const int hi   = lane >> 4;

  // A fragments: 16 query rows x K=128 (4 chunks of 32), resident in VGPRs
  Frag16 a[4];
  const _Float16* qrow = qf + (size_t)(qtile * 16 + l16) * CC;
#pragma unroll
  for (int c = 0; c < 4; ++c) {
    const _Float16* ap = qrow + c * 32 + hi * 16;
    a[c].u4[0] = *(const uint4*)(ap);
    a[c].u4[1] = *(const uint4*)(ap + 8);
  }

  float loc[8][NK];
#pragma unroll
  for (int v = 0; v < 8; ++v)
#pragma unroll
    for (int k = 0; k < NK; ++k) loc[v][k] = 0.0f;

  const int dvalid = LD - di;         // Ld - k + 1

  for (int t = wid; t < LD / 16; t += 4) {
    const int n = t * 16 + l16;       // doc position = accumulator N column
    const _Float16* drow = df + (size_t)n * CC;
    v8f acc = {};
#pragma unroll
    for (int c = 0; c < 4; ++c) {
      Frag16 bf;
      const _Float16* bp = drow + c * 32 + hi * 16;
      bf.u4[0] = ((const uint4*)bp)[0];
      bf.u4[1] = ((const uint4*)bp)[1];
      acc = wmma_f16(a[c].h, bf.h, acc);
    }
    const float mdv = (n < dvalid) ? p.dwm[b * LD + n] : 0.0f;
#pragma unroll
    for (int v = 0; v < 8; ++v) {
      float s = acc[v];
#pragma unroll
      for (int k = 0; k < NK; ++k) {
        float d = s - MU[k];
        loc[v][k] = fmaf(mdv, __expf(d * d * NC[k]), loc[v][k]);
      }
    }
  }

  // reduce per (query-row, kernel) into LDS
#pragma unroll
  for (int v = 0; v < 8; ++v) {
    int M = v + hi * 8;
#pragma unroll
    for (int k = 0; k < NK; ++k) atomicAdd(&ps[M * NK + k], loc[v][k]);
  }
  __syncthreads();

  const int qvalid = LQ - qi;         // Lq - k + 1
  for (int i = threadIdx.x; i < 16 * NK; i += 128) {
    int qr = i / NK, k = i % NK;
    int qpos = qtile * 16 + qr;
    float mq = (qpos < qvalid) ? p.qwm[b * LQ + qpos] : 0.0f;
    float lp = __logf(fmaxf(ps[i], 1e-10f)) * 0.01f * mq;
    atomicAdd(&p.pool[(size_t)b * 99 + combo * NK + k], lp);
  }
}

// ---------------------------------------------------------------------------
// dense + tanh head
// ---------------------------------------------------------------------------
__global__ void dense_tanh_kernel(const float* __restrict__ pool,
                                  const float* __restrict__ Wd,
                                  const float* __restrict__ bd,
                                  float* __restrict__ out) {
  int b = threadIdx.x;
  if (b < B_) {
    float s = bd[0];
    for (int i = 0; i < 99; ++i) s += pool[b * 99 + i] * Wd[i];
    out[b] = tanhf(s);
  }
}

// ---------------------------------------------------------------------------
// host launcher
// ---------------------------------------------------------------------------
static inline size_t align256(size_t x) { return (x + 255) & ~(size_t)255; }

extern "C" void kernel_launch(void* const* d_in, const int* in_sizes, int n_in,
                              void* d_out, int out_size, void* d_ws, size_t ws_size,
                              hipStream_t stream) {
  const float* qwt = (const float*)d_in[0];
  const float* dwt = (const float*)d_in[1];
  const float* qwm = (const float*)d_in[2];
  const float* dwm = (const float*)d_in[3];
  const float* Wu  = (const float*)d_in[4];
  const float* bu  = (const float*)d_in[5];
  const float* Wb  = (const float*)d_in[6];
  const float* bb  = (const float*)d_in[7];
  const float* Wt  = (const float*)d_in[8];
  const float* bt  = (const float*)d_in[9];
  const float* Wd  = (const float*)d_in[10];
  const float* bd  = (const float*)d_in[11];
  float* out = (float*)d_out;

  char* ws = (char*)d_ws;
  size_t off = 0;
  auto take = [&](size_t bytes) -> void* {
    void* pp = ws + off;
    off = align256(off + bytes);
    return pp;
  };
  const int QLA = LQ + 2, DLA = LD + 2;   // guard rows for tri-grams
  _Float16* qx   = (_Float16*)take((size_t)B_ * QLA * DP * 2);
  _Float16* dx   = (_Float16*)take((size_t)B_ * DLA * DP * 2);
  _Float16* w16u = (_Float16*)take((size_t)CC * 1 * DP * 2);
  _Float16* w16b = (_Float16*)take((size_t)CC * 2 * DP * 2);
  _Float16* w16t = (_Float16*)take((size_t)CC * 3 * DP * 2);
  _Float16* qfu  = (_Float16*)take((size_t)B_ * LQ * CC * 2);
  _Float16* qfb  = (_Float16*)take((size_t)B_ * LQ * CC * 2);
  _Float16* qft  = (_Float16*)take((size_t)B_ * LQ * CC * 2);
  _Float16* dfu  = (_Float16*)take((size_t)B_ * LD * CC * 2);
  _Float16* dfb  = (_Float16*)take((size_t)B_ * LD * CC * 2);
  _Float16* dft  = (_Float16*)take((size_t)B_ * LD * CC * 2);
  float*    pool = (float*)take((size_t)B_ * 99 * 4);

  // prep
  zero_f32_kernel<<<(B_ * 99 + 255) / 256, 256, 0, stream>>>(pool, B_ * 99);
  {
    size_t tq = (size_t)B_ * QLA * DP;
    cvt_x_kernel<<<(unsigned)((tq + 255) / 256), 256, 0, stream>>>(qwt, qx, LQ, QLA);
    size_t td = (size_t)B_ * DLA * DP;
    cvt_x_kernel<<<(unsigned)((td + 255) / 256), 256, 0, stream>>>(dwt, dx, LD, DLA);
  }
  cvt_w_kernel<<<(CC * 1 * DP + 255) / 256, 256, 0, stream>>>(Wu, w16u, 1);
  cvt_w_kernel<<<(CC * 2 * DP + 255) / 256, 256, 0, stream>>>(Wb, w16b, 2);
  cvt_w_kernel<<<(CC * 3 * DP + 255) / 256, 256, 0, stream>>>(Wt, w16t, 3);

  // n-gram convolutions (WMMA, async LDS staging)
  auto shA = [](int K) { return (size_t)(16 * K * 2 + 64); };
  ngram_conv_kernel<<<dim3(B_, LQ / 16), 128, shA(320), stream>>>(qx, w16u, bu, qfu, QLA, LQ, 320);
  ngram_conv_kernel<<<dim3(B_, LQ / 16), 128, shA(640), stream>>>(qx, w16b, bb, qfb, QLA, LQ, 640);
  ngram_conv_kernel<<<dim3(B_, LQ / 16), 128, shA(960), stream>>>(qx, w16t, bt, qft, QLA, LQ, 960);
  ngram_conv_kernel<<<dim3(B_, LD / 16), 128, shA(320), stream>>>(dx, w16u, bu, dfu, DLA, LD, 320);
  ngram_conv_kernel<<<dim3(B_, LD / 16), 128, shA(640), stream>>>(dx, w16b, bb, dfb, DLA, LD, 640);
  ngram_conv_kernel<<<dim3(B_, LD / 16), 128, shA(960), stream>>>(dx, w16t, bt, dft, DLA, LD, 960);

  // 9 intersections + RBF pooling (WMMA + fused exp)
  SimParams sp;
  sp.qf[0] = qfu; sp.qf[1] = qfb; sp.qf[2] = qft;
  sp.df[0] = dfu; sp.df[1] = dfb; sp.df[2] = dft;
  sp.qwm = qwm; sp.dwm = dwm; sp.pool = pool;
  sim_pool_kernel<<<dim3(B_, 9, 2), 128, 0, stream>>>(sp);

  // head
  dense_tanh_kernel<<<1, 64, 0, stream>>>(pool, Wd, bd, out);
  (void)in_sizes; (void)n_in; (void)out_size; (void)ws_size;
}
